// DynamicConv_73959336837221
// MI455X (gfx1250) — compile-verified, matrix operand order checked by reading, not run on hardware
//
#include <hip/hip_runtime.h>
#include <hip/hip_bf16.h>

typedef __bf16 v16bf __attribute__((ext_vector_type(16)));
typedef float  v8f   __attribute__((ext_vector_type(8)));

#define B_SZ   32
#define C_IN   256
#define C_OUT  256
#define F_DIM  64
#define T_DIM  64
#define NSPAT  (F_DIM * T_DIM)   // 4096
#define XT_H   66                 // 64 + halo rows
#define XT_W   66                 // 64 + halo cols (i-inner dim handles alignment)
#define AGGW_ELEMS ((size_t)B_SZ * 9 * C_OUT * C_IN)            // 18,874,368
#define XT_ELEMS   ((size_t)B_SZ * XT_H * XT_W * C_IN)          // 35,684,352
#define LDS_PITCH  72              // 64 ch + 8 pad: 144B rows, 16B aligned

// ---------------------------------------------------------------------------
// Kernel 1: attention = softmax((g @ dense_w + b) / 30) per batch sample
// ---------------------------------------------------------------------------
__global__ void attn_kernel(const float* __restrict__ g,
                            const float* __restrict__ dw,
                            const float* __restrict__ db,
                            float* __restrict__ att) {
    __shared__ float logits[B_SZ * 4];
    int t = threadIdx.x;
    if (t < B_SZ * 4) {
        int bI = t >> 2, k = t & 3;
        float s = db[k];
        for (int c = 0; c < 256; ++c)
            s += g[bI * 256 + c] * dw[c * 4 + k];
        logits[t] = s;
    }
    __syncthreads();
    if (t < B_SZ) {
        float l0 = logits[t*4+0], l1 = logits[t*4+1];
        float l2 = logits[t*4+2], l3 = logits[t*4+3];
        float m  = fmaxf(fmaxf(l0, l1), fmaxf(l2, l3));
        float e0 = __expf((l0 - m) / 30.f), e1 = __expf((l1 - m) / 30.f);
        float e2 = __expf((l2 - m) / 30.f), e3 = __expf((l3 - m) / 30.f);
        float inv = 1.f / (e0 + e1 + e2 + e3);
        att[t*4+0] = e0 * inv;  att[t*4+1] = e1 * inv;
        att[t*4+2] = e2 * inv;  att[t*4+3] = e3 * inv;
    }
}

// ---------------------------------------------------------------------------
// Kernel 2: per-sample mixed weights -> bf16, layout [b][kpos][o][i]
// ---------------------------------------------------------------------------
__global__ void aggw_kernel(const float* __restrict__ w,
                            const float* __restrict__ att,
                            unsigned short* __restrict__ aggw_u) {
    __bf16* aggw = (__bf16*)aggw_u;
    int idx = blockIdx.x * 256 + threadIdx.x;   // ((b*9+kp)<<16) | (o<<8) | i
    int i  = idx & 255;
    int o  = (idx >> 8) & 255;
    int r  = idx >> 16;
    int kp = r % 9;
    int b  = r / 9;
    float a0 = att[b*4+0], a1 = att[b*4+1], a2 = att[b*4+2], a3 = att[b*4+3];
    const int KSTRIDE = C_OUT * C_IN * 9;       // stride between experts
    int base = (o * 256 + i) * 9 + kp;
    float s = a0 * w[base]
            + a1 * w[base + KSTRIDE]
            + a2 * w[base + 2 * KSTRIDE]
            + a3 * w[base + 3 * KSTRIDE];
    aggw[idx] = (__bf16)s;
}

// ---------------------------------------------------------------------------
// Kernel 3: x (fp32, NCHW) -> bf16 transposed + zero halo:
//           xT[b][row(66)][col(66)][i(256)]  (spatial-major, channel-inner)
// ---------------------------------------------------------------------------
__global__ void xtrans_kernel(const float* __restrict__ x,
                              unsigned short* __restrict__ xT_u) {
    __bf16* xT = (__bf16*)xT_u;
    int idx = blockIdx.x * 256 + threadIdx.x;   // ((b*66+row)*66+col)*256 + i
    int i   = idx & 255;
    int t1  = idx >> 8;
    int col = t1 % XT_W;
    int t2  = t1 / XT_W;
    int row = t2 % XT_H;
    int b   = t2 / XT_H;
    float v = 0.f;
    if (row >= 1 && row <= 64 && col >= 1 && col <= 64) {
        v = x[(((size_t)b * 256 + i) * 64 + (row - 1)) * 64 + (col - 1)];
    }
    xT[idx] = (__bf16)v;
}

// ---------------------------------------------------------------------------
// Kernel 4: implicit-GEMM dynamic conv via v_wmma_f32_16x16x32_bf16
// Block: (b, 128 c_out, 128 spatial = 2 rows x 64 cols), 8 waves (2M x 4N)
// Wave: 64x32 output tile = 4x2 WMMA accumulators
// 36 stages of K=64: it = kp*4 + h (channel quarter innermost); each stage:
//   8 global_load_b128 + 8 ds_store_b128 + 24 ds_load_b128 + 16 WMMAs,
//   software-pipelined so next stage's global loads overlap the WMMAs.
// ---------------------------------------------------------------------------
__global__ __launch_bounds__(256)
void dynconv_wmma(const unsigned short* __restrict__ aggw_u,
                  const unsigned short* __restrict__ xT_u,
                  float* __restrict__ out) {
    const __bf16* aggw = (const __bf16*)aggw_u;
    const __bf16* xT   = (const __bf16*)xT_u;

    // 128 rows x 64 ch (+8 pad): 144B row stride => 16B-aligned b128 ops,
    // 36-dword bank stride => conflict-free across 16 lanes
    __shared__ __bf16 Alds[128 * LDS_PITCH];   // [o_local][i_local]
    __shared__ __bf16 Blds[128 * LDS_PITCH];   // [p_local][i_local]

    const int tid   = threadIdx.x;
    const int lane  = tid & 31;
    const int wave  = tid >> 5;
    const int waveM = wave >> 2;        // 0..1
    const int waveN = wave & 3;         // 0..3
    const int lrow  = lane & 15;        // A row / B col / C col within 16-tile
    const int lkh   = lane >> 4;        // K half (frags) / M half (C)

    const int b  = blockIdx.z;
    const int m0 = blockIdx.y * 128;    // c_out tile base
    const int p0 = blockIdx.x * 128;    // spatial tile base (2 rows)
    const int y0 = p0 >> 6;

    // staging maps (each thread moves two contiguous 32B bf16 vectors per tile)
    const int aOL = tid >> 1;           // A: o row 0..127
    const int aIH = tid & 1;            // A: which 16-i half of a 32-chunk
    const int bP  = tid & 127;          // B: spatial 0..127
    const int bIH = tid >> 7;           // B: which 16-i half of a 32-chunk
    const int bY  = y0 + (bP >> 6);     // unpadded row of this spatial pos
    const int bX  = bP & 63;            // unpadded col

    const __bf16* aggb = aggw + (size_t)b * (9 * C_OUT * C_IN)
                              + (size_t)(m0 + aOL) * 256 + aIH * 16;
    const __bf16* xTb  = xT + (size_t)b * (XT_H * XT_W * C_IN) + bIH * 16;

    v8f acc[4][2] = {};

    // stage it = kp*4 + h  (h = which 64-channel quarter)
    auto srcA = [&](int it) -> const __bf16* {
        int kp = it >> 2, h = it & 3;
        return aggb + (size_t)kp * (C_OUT * C_IN) + h * 64;
    };
    auto srcB = [&](int it) -> const __bf16* {
        int kp = it >> 2, h = it & 3;
        int dy = kp / 3 - 1, dx = kp % 3 - 1;
        return xTb + ((size_t)(bY + dy + 1) * XT_W + (bX + dx + 1)) * C_IN + h * 64;
    };

    v16bf va[2], vb[2];
    {
        const __bf16* pa = srcA(0);
        const __bf16* pb = srcB(0);
        va[0] = *(const v16bf*)pa;        va[1] = *(const v16bf*)(pa + 32);
        vb[0] = *(const v16bf*)pb;        vb[1] = *(const v16bf*)(pb + 32);
    }

    #pragma unroll 1
    for (int it = 0; it < 36; ++it) {
        __syncthreads();                    // previous compute done reading LDS
        *(v16bf*)&Alds[aOL * LDS_PITCH      + aIH * 16] = va[0];
        *(v16bf*)&Alds[aOL * LDS_PITCH + 32 + aIH * 16] = va[1];
        *(v16bf*)&Blds[bP  * LDS_PITCH      + bIH * 16] = vb[0];
        *(v16bf*)&Blds[bP  * LDS_PITCH + 32 + bIH * 16] = vb[1];

        // preload next stage (clamped on last iter); overlaps the 16 WMMAs
        int nit = (it < 35) ? it + 1 : it;
        {
            const __bf16* pa = srcA(nit);
            const __bf16* pb = srcB(nit);
            va[0] = *(const v16bf*)pa;    va[1] = *(const v16bf*)(pa + 32);
            vb[0] = *(const v16bf*)pb;    vb[1] = *(const v16bf*)(pb + 32);
        }
        __builtin_prefetch(srcB(it < 34 ? it + 2 : it), 0, 0);

        __syncthreads();                    // tiles visible
        #pragma unroll
        for (int kk = 0; kk < 2; ++kk) {    // two K=32 sub-steps
            v16bf afr[4], bfr[2];
            #pragma unroll
            for (int mi = 0; mi < 4; ++mi)
                afr[mi] = *(const v16bf*)
                    &Alds[(waveM * 64 + mi * 16 + lrow) * LDS_PITCH + kk * 32 + lkh * 16];
            #pragma unroll
            for (int ni = 0; ni < 2; ++ni)
                bfr[ni] = *(const v16bf*)
                    &Blds[(waveN * 32 + ni * 16 + lrow) * LDS_PITCH + kk * 32 + lkh * 16];
            #pragma unroll
            for (int mi = 0; mi < 4; ++mi)
                #pragma unroll
                for (int ni = 0; ni < 2; ++ni)
                    acc[mi][ni] = __builtin_amdgcn_wmma_f32_16x16x32_bf16(
                        false, afr[mi], false, bfr[ni],
                        (short)0, acc[mi][ni], false, false);
        }
    }

    // ---- epilogue: C/D layout => VGPR r, lane L: M = 8*(L>>4)+r, N = L&15
    float* ob = out + (size_t)b * C_OUT * NSPAT;
    #pragma unroll
    for (int mi = 0; mi < 4; ++mi)
        #pragma unroll
        for (int ni = 0; ni < 2; ++ni)
            #pragma unroll
            for (int r = 0; r < 8; ++r) {
                int o = m0 + waveM * 64 + mi * 16 + lkh * 8 + r;
                int p = p0 + waveN * 32 + ni * 16 + lrow;
                ob[(size_t)o * NSPAT + p] = acc[mi][ni][r];
            }
}

// ---------------------------------------------------------------------------
extern "C" void kernel_launch(void* const* d_in, const int* in_sizes, int n_in,
                              void* d_out, int out_size, void* d_ws, size_t ws_size,
                              hipStream_t stream) {
    const float* x  = (const float*)d_in[0];   // (32,256,64,64)
    const float* g  = (const float*)d_in[1];   // (32,256)
    const float* w  = (const float*)d_in[2];   // (4,256,256,3,3)
    const float* dw = (const float*)d_in[3];   // (256,4)
    const float* db = (const float*)d_in[4];   // (4,)
    float* out = (float*)d_out;                // (32,256,64,64)

    // workspace layout
    char* ws = (char*)d_ws;
    float*          att  = (float*)ws;                                   // 512 B
    unsigned short* aggw = (unsigned short*)(ws + 512);                  // 37.75 MB
    unsigned short* xT   = (unsigned short*)(ws + 512 + AGGW_ELEMS * 2); // 71.37 MB

    attn_kernel<<<1, 128, 0, stream>>>(g, dw, db, att);
    aggw_kernel<<<(int)(AGGW_ELEMS / 256), 256, 0, stream>>>(w, att, aggw);
    xtrans_kernel<<<(int)(XT_ELEMS / 256), 256, 0, stream>>>(x, xT);
    dynconv_wmma<<<dim3(NSPAT / 128, C_OUT / 128, B_SZ), 256, 0, stream>>>(aggw, xT, out);
}
